// T5Attention_39676907882560
// MI455X (gfx1250) — matrix operand order repaired and checked by reading, not compiled, not run on hardware
//
#include <hip/hip_runtime.h>
#include <stdint.h>

typedef __bf16 bf16_t;
typedef __attribute__((ext_vector_type(16))) __bf16 v16bf;
typedef __attribute__((ext_vector_type(8)))  float  v8f;

#define SLEN  2048
#define DHEAD 64
#define KBLK  64
#define NWAVE 4
#define QT    2      // query tiles (of 16) per wave
#define PROW  (KBLK + 8)

// Butterfly xor within each 16-lane row: DPP16 ROW_XMASK (0x160|mask).
#define ROWXOR(x, m)                                                          \
    __builtin_bit_cast(float, __builtin_amdgcn_mov_dpp(                       \
        __builtin_bit_cast(int, (x)), 0x160 | (m), 0xf, 0xf, true))

__global__ __launch_bounds__(128) void fattn_kernel(
    const float* __restrict__ Qp, const float* __restrict__ Kp,
    const float* __restrict__ Vp, const uint8_t* __restrict__ Mp,
    float* __restrict__ Op)
{
    // Row strides padded to 72 bf16 (= 36 dwords): 16 consecutive rows hit
    // distinct LDS bank groups (gcd(36,64)=4, cycle 16).
    __shared__ bf16_t Kt[KBLK][DHEAD + 8];            // [key][dim]
    __shared__ bf16_t Vt[DHEAD][KBLK + 8];            // [dim][key] transposed
    __shared__ bf16_t Pb[NWAVE][QT][16][PROW];        // per-wave/tile P staging
    __shared__ int    anyMaskSh;

    const int tid  = threadIdx.x;
    const int w    = tid >> 5;
    const int lane = tid & 31;
    const int half = lane >> 4;
    const int r    = lane & 15;

    const int qblocks = SLEN / (16 * QT * NWAVE);     // 16
    const int bh   = blockIdx.x / qblocks;            // 0..31
    const int qblk = blockIdx.x % qblocks;
    const int q0   = qblk * (16 * QT * NWAVE) + w * (16 * QT);
    const int bIdx = bh / 16;                         // batch index (H=16)

    // ---- Whole-batch mask scan (one uint4 per thread covers all 2048 bytes).
    // mask is (B,1,1,S): any True poisons the ENTIRE batch output in the
    // reference (post-softmax -inf fill -> every output element ±inf/nan),
    // so a single block-uniform flag suffices; hot loop sees one s_cbranch.
    if (tid == 0) anyMaskSh = 0;
    const uint4 mq = reinterpret_cast<const uint4*>(Mp + (size_t)bIdx * SLEN)[tid];
    __syncthreads();
    if ((mq.x | mq.y | mq.z | mq.w) != 0u) anyMaskSh = 1;   // benign race
    __syncthreads();
    const bool anyMask = (anyMaskSh != 0);

    // ---- Q as A-fragments for both tiles, prescaled by (1/sqrt(D))*log2(e)
    // so softmax runs in exp2 domain (bare v_exp_f32).
    const float qscale = 0.125f * 1.44269504088896340736f;
    v16bf a0[QT], a1[QT];
#pragma unroll
    for (int t = 0; t < QT; ++t) {
        const float* Qrow = Qp + ((size_t)bh * SLEN + q0 + t * 16 + r) * DHEAD;
#pragma unroll
        for (int e = 0; e < 8; ++e) {
            a0[t][e]     = (bf16_t)(Qrow[     half * 16 + e] * qscale);
            a0[t][e + 8] = (bf16_t)(Qrow[ 8 + half * 16 + e] * qscale);
            a1[t][e]     = (bf16_t)(Qrow[32 + half * 16 + e] * qscale);
            a1[t][e + 8] = (bf16_t)(Qrow[40 + half * 16 + e] * qscale);
        }
    }

    // Constant all-ones B fragment: P x ones = per-row sums (l accumulator)
    // via the matrix pipe instead of a DPP butterfly.
    v16bf bones;
#pragma unroll
    for (int e = 0; e < 16; ++e) bones[e] = (bf16_t)1.0f;

    // Per-wave/tile P staging pointers (write side: row j+half*8, col r).
    bf16_t* pbw[QT];
#pragma unroll
    for (int t = 0; t < QT; ++t) pbw[t] = &Pb[w][t][half * 8][r];

    // Online-softmax state (exp2 domain): slot j <-> query row (j + half*8).
    v8f mrow[QT], lrow[QT];
    v8f o[QT][4] = {};
#pragma unroll
    for (int t = 0; t < QT; ++t)
#pragma unroll
        for (int j = 0; j < 8; ++j) { mrow[t][j] = -__builtin_inff(); lrow[t][j] = 0.f; }

    const bf16_t ninf = (bf16_t)(-__builtin_inff());

    for (int kb = 0; kb < SLEN / KBLK; ++kb) {
        const int k0 = kb * KBLK;
        __syncthreads();                              // tiles free to overwrite
        const float* Kg = Kp + ((size_t)bh * SLEN + k0) * DHEAD;
        const float* Vg = Vp + ((size_t)bh * SLEN + k0) * DHEAD;
        const float4* Kg4 = reinterpret_cast<const float4*>(Kg);
        const float4* Vg4 = reinterpret_cast<const float4*>(Vg);
#pragma unroll
        for (int i = 0; i < 8; ++i) {                 // 4096 floats / (128thr*4)
            int e4  = tid + i * 128;
            int key = e4 >> 4;
            int d   = (e4 & 15) * 4;
            float4 kq = Kg4[e4];
            float4 vq = Vg4[e4];
            Kt[key][d + 0] = (bf16_t)kq.x;  Kt[key][d + 1] = (bf16_t)kq.y;
            Kt[key][d + 2] = (bf16_t)kq.z;  Kt[key][d + 3] = (bf16_t)kq.w;
            Vt[d + 0][key] = (bf16_t)vq.x;  Vt[d + 1][key] = (bf16_t)vq.y;
            Vt[d + 2][key] = (bf16_t)vq.z;  Vt[d + 3][key] = (bf16_t)vq.w;
        }
        if (kb + 1 < SLEN / KBLK) {                   // warm next tile
            __builtin_prefetch(Kg + KBLK * DHEAD + tid * 32, 0, 0);
            __builtin_prefetch(Vg + KBLK * DHEAD + tid * 32, 0, 0);
        }
        __syncthreads();

        // ---- QK^T: 4 col tiles x 2 K-slices x QT tiles = 16 WMMA.
        v8f s[QT][4];
#pragma unroll
        for (int nt = 0; nt < 4; ++nt) {
            v16bf bk0, bk1;
            const bf16_t* p0 = &Kt[nt * 16 + r][     half * 16];
            const bf16_t* p1 = &Kt[nt * 16 + r][32 + half * 16];
#pragma unroll
            for (int e = 0; e < 16; ++e) { bk0[e] = p0[e]; bk1[e] = p1[e]; }
#pragma unroll
            for (int t = 0; t < QT; ++t) {
                v8f acc = {};
                acc = __builtin_amdgcn_wmma_f32_16x16x32_bf16(false, a0[t], false, bk0, (short)0, acc, false, false);
                acc = __builtin_amdgcn_wmma_f32_16x16x32_bf16(false, a1[t], false, bk1, (short)0, acc, false, false);
                s[t][nt] = acc;
            }
        }

        v8f avv[QT];                                  // alpha per row (vector)
#pragma unroll
        for (int t = 0; t < QT; ++t) {
#pragma unroll
            for (int j = 0; j < 8; ++j) {
                float tmx = fmaxf(fmaxf(s[t][0][j], s[t][1][j]),
                                  fmaxf(s[t][2][j], s[t][3][j]));
                tmx = fmaxf(tmx, ROWXOR(tmx, 1));
                tmx = fmaxf(tmx, ROWXOR(tmx, 2));
                tmx = fmaxf(tmx, ROWXOR(tmx, 4));
                tmx = fmaxf(tmx, ROWXOR(tmx, 8));
                float mn = fmaxf(mrow[t][j], tmx);
                avv[t][j] = __builtin_amdgcn_exp2f(mrow[t][j] - mn);
                mrow[t][j] = mn;
                bf16_t* pr = pbw[t] + j * PROW;
                pr[ 0] = (bf16_t)__builtin_amdgcn_exp2f(s[t][0][j] - mn);
                pr[16] = (bf16_t)__builtin_amdgcn_exp2f(s[t][1][j] - mn);
                pr[32] = (bf16_t)__builtin_amdgcn_exp2f(s[t][2][j] - mn);
                pr[48] = (bf16_t)__builtin_amdgcn_exp2f(s[t][3][j] - mn);
            }
            // vector alpha-rescale of accumulators (packed-f32)
            o[t][0] *= avv[t]; o[t][1] *= avv[t];
            o[t][2] *= avv[t]; o[t][3] *= avv[t];
        }

        // Cold path, block-uniform branch (never taken for all-False mask):
        // any masked key => reference output for this batch is ±inf/nan
        // everywhere, so poisoning staged P reproduces "garbage" semantics
        // without any per-lane EXEC manipulation in the hot path.
        if (anyMask) {
#pragma unroll
            for (int t = 0; t < QT; ++t)
#pragma unroll
                for (int j = 0; j < 8; ++j) {
                    bf16_t* pr = pbw[t] + j * PROW;
                    pr[0] = ninf; pr[16] = ninf; pr[32] = ninf; pr[48] = ninf;
                }
        }

        // intra-wave LDS visibility before re-reading P in A-fragment layout
        asm volatile("s_wait_dscnt 0" ::: "memory");

        v16bf pa0[QT], pa1[QT];                       // P as A: keys 0..31 / 32..63
#pragma unroll
        for (int t = 0; t < QT; ++t) {
            const bf16_t* c0 = &Pb[w][t][r][     half * 16];
            const bf16_t* c1 = &Pb[w][t][r][ 8 + half * 16];
            const bf16_t* c2 = &Pb[w][t][r][32 + half * 16];
            const bf16_t* c3 = &Pb[w][t][r][40 + half * 16];
#pragma unroll
            for (int e = 0; e < 8; ++e) {
                pa0[t][e] = c0[e]; pa0[t][e + 8] = c1[e];
                pa1[t][e] = c2[e]; pa1[t][e + 8] = c3[e];
            }
        }

        // Row sums of staged P via matrix pipe; slot j == sum of row (j+half*8).
#pragma unroll
        for (int t = 0; t < QT; ++t) {
            v8f rs = {};
            rs = __builtin_amdgcn_wmma_f32_16x16x32_bf16(false, pa0[t], false, bones, (short)0, rs, false, false);
            rs = __builtin_amdgcn_wmma_f32_16x16x32_bf16(false, pa1[t], false, bones, (short)0, rs, false, false);
            lrow[t] = lrow[t] * avv[t] + rs;          // vector fma (packed)
        }

        // O += P(16x64) * V(64x64): 4 dout tiles x 2 key-slices x QT = 16 WMMA.
#pragma unroll
        for (int nt = 0; nt < 4; ++nt) {
            v16bf bv0, bv1;
            const bf16_t* v0p = &Vt[nt * 16 + r][     half * 16];
            const bf16_t* v1p = &Vt[nt * 16 + r][32 + half * 16];
#pragma unroll
            for (int e = 0; e < 16; ++e) { bv0[e] = v0p[e]; bv1[e] = v1p[e]; }
#pragma unroll
            for (int t = 0; t < QT; ++t) {
                o[t][nt] = __builtin_amdgcn_wmma_f32_16x16x32_bf16(false, pa0[t], false, bv0, (short)0, o[t][nt], false, false);
                o[t][nt] = __builtin_amdgcn_wmma_f32_16x16x32_bf16(false, pa1[t], false, bv1, (short)0, o[t][nt], false, false);
            }
        }
    }

    // ---- epilogue: normalize and store (C layout: slot j -> row j+half*8, col r)
#pragma unroll
    for (int t = 0; t < QT; ++t) {
        v8f linv;
#pragma unroll
        for (int j = 0; j < 8; ++j) linv[j] = 1.0f / lrow[t][j];
#pragma unroll
        for (int nt = 0; nt < 4; ++nt) {
            v8f ov = o[t][nt] * linv;                 // vector (packed)
#pragma unroll
            for (int j = 0; j < 8; ++j) {
                const int qrow = q0 + t * 16 + j + half * 8;
                const int dcol = nt * 16 + r;
                Op[((size_t)bh * SLEN + qrow) * DHEAD + dcol] = ov[j];
            }
        }
    }
}

extern "C" void kernel_launch(void* const* d_in, const int* in_sizes, int n_in,
                              void* d_out, int out_size, void* d_ws, size_t ws_size,
                              hipStream_t stream) {
    const float*   x    = (const float*)d_in[0];
    const float*   k    = (const float*)d_in[1];
    const float*   v    = (const float*)d_in[2];
    const uint8_t* mask = (const uint8_t*)d_in[3];
    float* out = (float*)d_out;

    dim3 grid(2 * 16 * (SLEN / (16 * QT * NWAVE)));  // B*H * 16 = 512 WGs
    dim3 block(128);                                  // 4 wave32 per WG
    hipLaunchKernelGGL(fattn_kernel, grid, block, 0, stream, x, k, v, mask, out);
}
